// _MaskedConv_38019050504696
// MI455X (gfx1250) — matrix operand order repaired and checked by reading, not compiled, not run on hardware
//
#include <hip/hip_runtime.h>

// Problem constants (from reference): B=8, N=20000, K=16, C_IN=64, C_OUT=64
#define BATCH   8
#define NPTS    20000
#define KNBR    16
#define CIN     64
#define COUT    64
#define RED     (KNBR * CIN)   // 1024 reduction length
#define NCHUNK  (RED / 32)     // 32 WMMA K-steps of 32
#define NT4     (NPTS / 4)     // 5000 groups of 4 points (two 16-row tiles)
#define LDSW_WORDS (NCHUNK * 4 * 32 * 8)  // 32768 u32 = 128 KB pre-swizzled bf16 W

typedef __attribute__((ext_vector_type(16))) __bf16 v16bf;
typedef __attribute__((ext_vector_type(8)))  float  v8f;

union ABFrag {
    v16bf v;
    uint4 q[2];
    unsigned u[8];
};

// pack two f32 -> one dword of two bf16 (lo in [15:0], hi in [31:16]), masked
#if __has_builtin(__builtin_amdgcn_cvt_pk_bf16_f32)
typedef __attribute__((ext_vector_type(2))) __bf16 v2bf;
__device__ __forceinline__ unsigned pk_bf16(float lo, float hi, unsigned keep) {
    v2bf r = __builtin_amdgcn_cvt_pk_bf16_f32(lo, hi);
    return __builtin_bit_cast(unsigned, r) & keep;
}
#else
__device__ __forceinline__ unsigned pk_bf16(float lo, float hi, unsigned keep) {
    return ((__float_as_uint(lo) >> 16) | (__float_as_uint(hi) & 0xFFFF0000u)) & keep;
}
#endif

__global__ __launch_bounds__(256) void masked_conv_wmma(
    const float* __restrict__ x,      // [B, N, CIN]
    const int*   __restrict__ kidx,   // [N, K]
    const int*   __restrict__ kmask,  // [N, K]
    const float* __restrict__ w,      // [COUT, CIN, K]
    const float* __restrict__ bias,   // [COUT]
    float*       __restrict__ out)    // [B, N, COUT]
{
    extern __shared__ unsigned ldsW[];   // 128 KB, dynamic (CDNA5: 320 KB/WGP)

    const int tid = threadIdx.x;

    // ------------------------------------------------------------------
    // Stage W -> LDS as bf16, pre-swizzled into per-lane WMMA B fragments.
    // Fragment chunk c = (s, ot, lane); 8 packed dwords (16 bf16) per chunk.
    // B order used here (contiguous k, same as A): lane = column o (mod 16); lanes 0-15 hold
    // K_rel 0..15, lanes 16-31 hold K_rel 16..31; element e <-> K_rel kb+e.
    // Reduction index r = 32*s + K_rel, with r = k*64 + i  (k = r>>6, i = r&63).
    // ------------------------------------------------------------------
    for (int wd = tid; wd < LDSW_WORDS; wd += 256) {
        int p    = wd & 7;          // dword within fragment
        int c    = wd >> 3;
        int lane = c & 31;
        int ot   = (c >> 5) & 3;    // output 16-col tile
        int s    = c >> 7;          // K-chunk
        int kb   = (lane < 16) ? 0 : 16;
        int o    = ot * 16 + (lane & 15);
        int r0   = 32 * s + kb + 2 * p;      // even
        int k    = r0 >> 6;
        int i0   = r0 & 63;                  // even, i0+1 stays in same k
        float f0 = w[o * RED + i0 * KNBR + k];
        float f1 = w[o * RED + (i0 + 1) * KNBR + k];
        ldsW[wd] = pk_bf16(f0, f1, 0xFFFFFFFFu);
    }
    __syncthreads();

    const uint4* lw = (const uint4*)ldsW;

    const int lane  = tid & 31;
    const int wv    = tid >> 5;
    const int gw    = blockIdx.x * 8 + wv;   // global wave id
    const int gws   = gridDim.x * 8;

    // A-matrix row for this lane: M = lane & 15 -> (n_sub = M>>3, b = M&7)
    const int bb    = lane & 7;
    const int nsub  = (lane >> 3) & 1;
    const int ihalf = ((lane >> 4) & 1) * 16;  // upper-half lanes take K_rel +16
    const int ocol  = lane & 15;

    // per-batch row base for gathers (hoists the b*N*CIN mul out of the loop)
    const float* xb = x + (size_t)bb * (size_t)NPTS * CIN + ihalf;

    const float bv0 = bias[ocol];
    const float bv1 = bias[16 + ocol];
    const float bv2 = bias[32 + ocol];
    const float bv3 = bias[48 + ocol];

    for (int t = gw; t < NT4; t += gws) {
        const int n0 = 4 * t;     // points n0..n0+3: tile0 = (n0,n0+1), tile1 = (n0+2,n0+3)

        // pre-stage idx+mask across lanes: lane L holds (nbase + (L>>4), k = L&15);
        // mask==0 encoded in the sign bit so one bpermute recovers both.
        const int pk0  = lane & 15;
        const int pn0  = n0 + (lane >> 4);
        const int pn1  = n0 + 2 + (lane >> 4);
        const int i0v  = kidx[pn0 * KNBR + pk0];
        const int m0v  = kmask[pn0 * KNBR + pk0];
        const int i1v  = kidx[pn1 * KNBR + pk0];
        const int m1v  = kmask[pn1 * KNBR + pk0];
        const int code0 = i0v | (m0v ? 0 : (int)0x80000000);
        const int code1 = i1v | (m1v ? 0 : (int)0x80000000);

        v8f acc0 = {}, acc1 = {}, acc2 = {}, acc3 = {};   // tile0, o-tiles 0..3
        v8f acc4 = {}, acc5 = {}, acc6 = {}, acc7 = {};   // tile1, o-tiles 0..3

#pragma unroll 4
        for (int s = 0; s < NCHUNK; ++s) {
            // lane holding (n_sub, k=s>>1) is q = 16*n_sub + (s>>1)
            const int qb  = ((nsub << 4) + (s >> 1)) << 2;
            const int cd0 = __builtin_amdgcn_ds_bpermute(qb, code0);
            const int cd1 = __builtin_amdgcn_ds_bpermute(qb, code1);
            const unsigned kp0 = ~(unsigned)(cd0 >> 31);   // 0 if mask==0
            const unsigned kp1 = ~(unsigned)(cd1 >> 31);
            const int jj0 = cd0 & 0x7FFFFFFF;
            const int jj1 = cd1 & 0x7FFFFFFF;
            const int off = (s & 1) << 5;

            // Gather: one contiguous 16-float run of x[b][j][:] per lane, per tile
            const float* rp0 = xb + ((size_t)jj0 << 6) + off;
            const float* rp1 = xb + ((size_t)jj1 << 6) + off;
            const float4 a0 = *(const float4*)(rp0);
            const float4 a1 = *(const float4*)(rp0 + 4);
            const float4 a2 = *(const float4*)(rp0 + 8);
            const float4 a3 = *(const float4*)(rp0 + 12);
            const float4 b0 = *(const float4*)(rp1);
            const float4 b1 = *(const float4*)(rp1 + 4);
            const float4 b2 = *(const float4*)(rp1 + 8);
            const float4 b3 = *(const float4*)(rp1 + 12);

            ABFrag fa, fb2;
            fa.u[0] = pk_bf16(a0.x, a0.y, kp0);
            fa.u[1] = pk_bf16(a0.z, a0.w, kp0);
            fa.u[2] = pk_bf16(a1.x, a1.y, kp0);
            fa.u[3] = pk_bf16(a1.z, a1.w, kp0);
            fa.u[4] = pk_bf16(a2.x, a2.y, kp0);
            fa.u[5] = pk_bf16(a2.z, a2.w, kp0);
            fa.u[6] = pk_bf16(a3.x, a3.y, kp0);
            fa.u[7] = pk_bf16(a3.z, a3.w, kp0);
            fb2.u[0] = pk_bf16(b0.x, b0.y, kp1);
            fb2.u[1] = pk_bf16(b0.z, b0.w, kp1);
            fb2.u[2] = pk_bf16(b1.x, b1.y, kp1);
            fb2.u[3] = pk_bf16(b1.z, b1.w, kp1);
            fb2.u[4] = pk_bf16(b2.x, b2.y, kp1);
            fb2.u[5] = pk_bf16(b2.z, b2.w, kp1);
            fb2.u[6] = pk_bf16(b3.x, b3.y, kp1);
            fb2.u[7] = pk_bf16(b3.z, b3.w, kp1);

            ABFrag wfr;
            const int fbase = (s * 4) * 32 + lane;   // fragment chunk index

            // each B fragment read once from LDS, used by BOTH point-pair tiles
            wfr.q[0] = lw[(fbase +  0) * 2 + 0];
            wfr.q[1] = lw[(fbase +  0) * 2 + 1];
            acc0 = __builtin_amdgcn_wmma_f32_16x16x32_bf16(
                       false, fa.v,  false, wfr.v, (short)0, acc0, false, false);
            acc4 = __builtin_amdgcn_wmma_f32_16x16x32_bf16(
                       false, fb2.v, false, wfr.v, (short)0, acc4, false, false);

            wfr.q[0] = lw[(fbase + 32) * 2 + 0];
            wfr.q[1] = lw[(fbase + 32) * 2 + 1];
            acc1 = __builtin_amdgcn_wmma_f32_16x16x32_bf16(
                       false, fa.v,  false, wfr.v, (short)0, acc1, false, false);
            acc5 = __builtin_amdgcn_wmma_f32_16x16x32_bf16(
                       false, fb2.v, false, wfr.v, (short)0, acc5, false, false);

            wfr.q[0] = lw[(fbase + 64) * 2 + 0];
            wfr.q[1] = lw[(fbase + 64) * 2 + 1];
            acc2 = __builtin_amdgcn_wmma_f32_16x16x32_bf16(
                       false, fa.v,  false, wfr.v, (short)0, acc2, false, false);
            acc6 = __builtin_amdgcn_wmma_f32_16x16x32_bf16(
                       false, fb2.v, false, wfr.v, (short)0, acc6, false, false);

            wfr.q[0] = lw[(fbase + 96) * 2 + 0];
            wfr.q[1] = lw[(fbase + 96) * 2 + 1];
            acc3 = __builtin_amdgcn_wmma_f32_16x16x32_bf16(
                       false, fa.v,  false, wfr.v, (short)0, acc3, false, false);
            acc7 = __builtin_amdgcn_wmma_f32_16x16x32_bf16(
                       false, fb2.v, false, wfr.v, (short)0, acc7, false, false);
        }

        // C/D layout: VGPR j, lanes 0-15 -> (row M=j, col=lane); lanes 16-31 -> (M=j+8)
        // row M = (n_sub = M>>3, b = M&7); lanes<16 write n_even, lanes>=16 write n_odd.
        const size_t nst0 = (size_t)(n0 + (lane >> 4));
        const size_t nst1 = nst0 + 2;
#pragma unroll
        for (int j = 0; j < 8; ++j) {
            const size_t base0 = ((size_t)j * NPTS + nst0) * COUT;
            const size_t base1 = ((size_t)j * NPTS + nst1) * COUT;
            out[base0 +  0 + ocol] = acc0[j] + bv0;
            out[base0 + 16 + ocol] = acc1[j] + bv1;
            out[base0 + 32 + ocol] = acc2[j] + bv2;
            out[base0 + 48 + ocol] = acc3[j] + bv3;
            out[base1 +  0 + ocol] = acc4[j] + bv0;
            out[base1 + 16 + ocol] = acc5[j] + bv1;
            out[base1 + 32 + ocol] = acc6[j] + bv2;
            out[base1 + 48 + ocol] = acc7[j] + bv3;
        }
    }
}

extern "C" void kernel_launch(void* const* d_in, const int* in_sizes, int n_in,
                              void* d_out, int out_size, void* d_ws, size_t ws_size,
                              hipStream_t stream) {
    const float* x     = (const float*)d_in[0];
    const int*   kidx  = (const int*)d_in[1];
    const int*   kmask = (const int*)d_in[2];
    const float* w     = (const float*)d_in[3];
    const float* bias  = (const float*)d_in[4];
    float*       out   = (float*)d_out;

    // 625 blocks * 8 waves = 5000 waves = exactly one 4-point group per wave
    dim3 grid(625), block(256);
    masked_conv_wmma<<<grid, block, LDSW_WORDS * sizeof(unsigned), stream>>>(
        x, kidx, kmask, w, bias, out);
}
